// MultiHeadSelfAttention_35545149342352
// MI455X (gfx1250) — compile-verified, matrix operand order checked by reading, not run on hardware
//
#include <hip/hip_runtime.h>
#include <hip/hip_bf16.h>

typedef __attribute__((ext_vector_type(16))) __bf16 v16bf;
typedef __attribute__((ext_vector_type(8)))  float  v8f;
typedef unsigned int u32;
typedef int v4i __attribute__((vector_size(16)));   // matches builtin param type

#define NUM_HEADS 16
#define SEQ 2048
#define EMB 1024
#define HD 64
#define BATCH 2
#define MROWS (BATCH * SEQ)   // 4096

#if defined(__HIP_DEVICE_COMPILE__) &&                                  \
    __has_builtin(__builtin_amdgcn_global_load_async_to_lds_b128) &&    \
    __has_builtin(__builtin_amdgcn_s_wait_asynccnt)
#define HAVE_ASYNC_LDS 1
#else
#define HAVE_ASYNC_LDS 0
#endif

union Frag {
    v16bf v;
    u32   u[8];
};

__device__ __forceinline__ int kpair_off(int lane, int p) {
    // ISA 7.12.2: 16-bit A/B fragment K mapping for dword p (holds K=k,k+1)
    return ((p < 4) ? 0 : 16) + ((lane >> 4) & 1) * 8 + 2 * (p & 3);
}

__device__ __forceinline__ __bf16 f2bf(float f) { return (__bf16)f; }

// 16-byte global -> LDS copy; async DMA path on gfx1250 when available
__device__ __forceinline__ void async_copy16(const void* g, void* l) {
#if HAVE_ASYNC_LDS
    __builtin_amdgcn_global_load_async_to_lds_b128(
        (__attribute__((address_space(1))) v4i*)(void*)g,
        (__attribute__((address_space(3))) v4i*)(void*)l, 0, 0);
#else
    *(uint4*)l = *(const uint4*)g;
#endif
}

__device__ __forceinline__ void async_wait() {
#if HAVE_ASYNC_LDS
    __builtin_amdgcn_s_wait_asynccnt(0);
#endif
}

// load an 8-dword (16 x bf16) WMMA fragment from a row-major row pointer
#define LOADF(dst, rowptr, koff)                                                   \
    do {                                                                           \
        _Pragma("unroll") for (int _p = 0; _p < 8; ++_p)                           \
            (dst).u[_p] = *(const u32*)((rowptr) + (koff) + kpair_off(lane, _p));  \
    } while (0)

#define WMMA_BF16(a, b, c)                                                         \
    __builtin_amdgcn_wmma_f32_16x16x32_bf16(false, (a), false, (b), (short)0, (c), \
                                            false, false)

// ---------------------------------------------------------------------------
// Kernel 1: fp32 -> bf16 cast (grid-stride)
// ---------------------------------------------------------------------------
__global__ void cast_f32_bf16(const float* __restrict__ in, __bf16* __restrict__ out,
                              int n) {
    for (int i = blockIdx.x * blockDim.x + threadIdx.x; i < n;
         i += gridDim.x * blockDim.x)
        out[i] = f2bf(in[i]);
}

// ---------------------------------------------------------------------------
// Kernel 2/4: bf16 GEMM  C[M,N] = A[M,K] * W[N,K]^T + bias
//   EPI==0: QKV epilogue -> scatter bf16 into Q/K/V [B*H][S][HD], Q scaled 1/8
//   EPI==1: fp32 output, row-major [M][N]
// Workgroup: 256 threads = 8 waves, tile 64(M) x 128(N); wave tile 32x32.
// ---------------------------------------------------------------------------
template <int EPI>
__global__ void gemm_bf16(const __bf16* __restrict__ A, const __bf16* __restrict__ W,
                          const float* __restrict__ bias, float* __restrict__ Cout,
                          __bf16* __restrict__ Qb, __bf16* __restrict__ Kb,
                          __bf16* __restrict__ Vb, int N) {
    constexpr int K = EMB;  // 1024
    __shared__ __align__(16) __bf16 As[64 * 32];
    __shared__ __align__(16) __bf16 Ws[128 * 32];

    const int tid = threadIdx.x;
    const int w = tid >> 5;
    const int lane = tid & 31;
    const int nBlocks = N >> 7;
    const int mBase = (blockIdx.x / nBlocks) * 64;
    const int nBase = (blockIdx.x % nBlocks) * 128;
    const int wm = w & 1;      // 0..1
    const int wn = w >> 1;     // 0..3

    v8f acc[2][2] = {};

    for (int kk = 0; kk < K; kk += 32) {
        __syncthreads();
        // stage A tile: 64 rows x 32 halves (= 256 chunks of 16B, 1/thread)
        {
            int r = tid >> 2, c = tid & 3;
            async_copy16(A + (size_t)(mBase + r) * K + kk + c * 8,
                         As + r * 32 + c * 8);
        }
        // stage W tile: 128 rows x 32 halves (= 512 chunks of 16B, 2/thread)
#pragma unroll
        for (int i = tid; i < 512; i += 256) {
            int r = i >> 2, c = i & 3;
            async_copy16(W + (size_t)(nBase + r) * K + kk + c * 8,
                         Ws + r * 32 + c * 8);
        }
        if (kk + 32 < K) {
            __builtin_prefetch(A + (size_t)(mBase + (tid >> 2)) * K + kk + 32, 0, 0);
            __builtin_prefetch(W + (size_t)(nBase + (tid >> 1)) * K + kk + 32, 0, 0);
        }
        async_wait();
        __syncthreads();

        Frag af[2], wf[2];
#pragma unroll
        for (int mt = 0; mt < 2; ++mt)
            LOADF(af[mt], As + (wm * 32 + mt * 16 + (lane & 15)) * 32, 0);
#pragma unroll
        for (int nt = 0; nt < 2; ++nt)
            LOADF(wf[nt], Ws + (wn * 32 + nt * 16 + (lane & 15)) * 32, 0);
#pragma unroll
        for (int mt = 0; mt < 2; ++mt)
#pragma unroll
            for (int nt = 0; nt < 2; ++nt)
                acc[mt][nt] = WMMA_BF16(af[mt].v, wf[nt].v, acc[mt][nt]);
    }

    const int hi = lane >> 4;
    const int nc = lane & 15;
#pragma unroll
    for (int mt = 0; mt < 2; ++mt)
#pragma unroll
        for (int nt = 0; nt < 2; ++nt)
#pragma unroll
            for (int r = 0; r < 8; ++r) {
                int row = mBase + wm * 32 + mt * 16 + r + 8 * hi;
                int col = nBase + wn * 32 + nt * 16 + nc;
                float val = acc[mt][nt][r] + bias[col];
                if (EPI == 1) {
                    Cout[(size_t)row * N + col] = val;
                } else {
                    int which = col >> 10;           // 0=Q,1=K,2=V
                    int h = (col >> 6) & 15;
                    int d = col & 63;
                    int b = row >> 11;
                    int s = row & 2047;
                    if (which == 0) val *= 0.125f;   // 1/sqrt(HD)
                    size_t idx = (((size_t)(b * NUM_HEADS + h)) * SEQ + s) * HD + d;
                    __bf16* dst = (which == 0) ? Qb : ((which == 1) ? Kb : Vb);
                    dst[idx] = f2bf(val);
                }
            }
}

// ---------------------------------------------------------------------------
// Kernel 3: flash attention over per-head bf16 Q/K/V [B*H][S][HD]
// grid = B*H*(S/128); block = 256 (8 waves); wave owns 16 q rows.
// KV processed 32 keys at a time; online softmax in f32.
// ---------------------------------------------------------------------------
__global__ void attn_fwd(const __bf16* __restrict__ Qb, const __bf16* __restrict__ Kb,
                         const __bf16* __restrict__ Vb, __bf16* __restrict__ Ob) {
    __shared__ __align__(16) __bf16 kbuf[32 * 64];  // K block row-major [key][d]
    __shared__ __align__(16) __bf16 vtb[64 * 32];   // V block transposed [d][key]
    __shared__ __align__(16) __bf16 pbuf[8][16 * 32];

    const int tid = threadIdx.x;
    const int w = tid >> 5;
    const int lane = tid & 31;
    const int qblk = blockIdx.x % (SEQ / 128);
    const int bh = blockIdx.x / (SEQ / 128);
    const size_t base = (size_t)bh * SEQ * HD;
    const int qrow = qblk * 128 + w * 16 + (lane & 15);

    // Q fragments: 16 rows x 64 K, split into two K=32 fragments
    Frag qa[2];
#pragma unroll
    for (int f = 0; f < 2; ++f)
#pragma unroll
        for (int p = 0; p < 8; ++p)
            qa[f].u[p] = *(const u32*)(Qb + base + (size_t)qrow * HD + f * 32 +
                                       kpair_off(lane, p));

    v8f o[4] = {};
    float mrun[8], lrun[8];
#pragma unroll
    for (int r = 0; r < 8; ++r) { mrun[r] = -3.0e38f; lrun[r] = 0.0f; }

    for (int kv = 0; kv < SEQ; kv += 32) {
        __syncthreads();
        // stage K block: 32 rows x 64 halves (= 256 chunks of 16B, 1/thread)
        {
            int key = tid >> 3, c = tid & 7;
            async_copy16(Kb + base + (size_t)(kv + key) * HD + c * 8,
                         kbuf + key * 64 + c * 8);
        }
        // stage V block transposed (coalesced global reads, scattered LDS b16)
        {
            const __bf16* vg = Vb + base + (size_t)kv * HD;
#pragma unroll
            for (int j = tid; j < 2048; j += 256) {
                int key = j >> 6, d = j & 63;
                vtb[d * 32 + key] = vg[j];
            }
        }
        async_wait();
        __syncthreads();

        // energy: 16 q-rows x 32 keys (two 16x16 tiles, K=64 contraction)
        v8f e0 = {}, e1 = {};
        {
            Frag kf;
            const __bf16* kr0 = kbuf + (lane & 15) * 64;         // keys 0..15
            LOADF(kf, kr0, 0);  e0 = WMMA_BF16(qa[0].v, kf.v, e0);
            LOADF(kf, kr0, 32); e0 = WMMA_BF16(qa[1].v, kf.v, e0);
            const __bf16* kr1 = kbuf + (16 + (lane & 15)) * 64;  // keys 16..31
            LOADF(kf, kr1, 0);  e1 = WMMA_BF16(qa[0].v, kf.v, e1);
            LOADF(kf, kr1, 32); e1 = WMMA_BF16(qa[1].v, kf.v, e1);
        }

        // online softmax (row = VGPR index r + 8*(lane>=16); 16 lanes hold a row)
        // reductions batched over all 8 rows so the 8 ds_bpermute per step
        // issue back-to-back instead of serializing on s_wait_dscnt 0
        float red[8], alpha[8];
#pragma unroll
        for (int r = 0; r < 8; ++r) red[r] = fmaxf(e0[r], e1[r]);
#pragma unroll
        for (int off = 8; off; off >>= 1)
#pragma unroll
            for (int r = 0; r < 8; ++r)
                red[r] = fmaxf(red[r], __shfl_xor(red[r], off, 32));
#pragma unroll
        for (int r = 0; r < 8; ++r) {
            float mnew = fmaxf(mrun[r], red[r]);
            alpha[r] = __expf(mrun[r] - mnew);
            float p0 = __expf(e0[r] - mnew);
            float p1 = __expf(e1[r] - mnew);
            e0[r] = p0;
            e1[r] = p1;
            mrun[r] = mnew;
            red[r] = p0 + p1;
        }
#pragma unroll
        for (int off = 8; off; off >>= 1)
#pragma unroll
            for (int r = 0; r < 8; ++r)
                red[r] += __shfl_xor(red[r], off, 32);
#pragma unroll
        for (int r = 0; r < 8; ++r) {
            lrun[r] = lrun[r] * alpha[r] + red[r];
#pragma unroll
            for (int t = 0; t < 4; ++t) o[t][r] *= alpha[r];
        }

        // C-layout -> A-layout for P via LDS
        const int hi = lane >> 4, nc = lane & 15;
        {
            __bf16* pw = pbuf[w];
#pragma unroll
            for (int r = 0; r < 8; ++r) {
                int m = r + 8 * hi;
                pw[m * 32 + nc] = f2bf(e0[r]);
                pw[m * 32 + 16 + nc] = f2bf(e1[r]);
            }
        }
        __syncthreads();

        Frag pf;
        LOADF(pf, pbuf[w] + (lane & 15) * 32, 0);
#pragma unroll
        for (int t = 0; t < 4; ++t) {
            Frag vf;
            LOADF(vf, vtb + (t * 16 + (lane & 15)) * 32, 0);
            o[t] = WMMA_BF16(pf.v, vf.v, o[t]);
        }
    }

    // normalize + store to [B][S][H*HD] bf16
    const int hi = lane >> 4, nc = lane & 15;
    const int b = bh >> 4, h = bh & 15;
#pragma unroll
    for (int r = 0; r < 8; ++r) {
        float inv = 1.0f / lrun[r];
        int m = r + 8 * hi;
        int srow = qblk * 128 + w * 16 + m;
#pragma unroll
        for (int t = 0; t < 4; ++t) {
            int d = t * 16 + nc;
            Ob[((size_t)b * SEQ + srow) * EMB + h * HD + d] = f2bf(o[t][r] * inv);
        }
    }
}

// ---------------------------------------------------------------------------
extern "C" void kernel_launch(void* const* d_in, const int* in_sizes, int n_in,
                              void* d_out, int out_size, void* d_ws, size_t ws_size,
                              hipStream_t stream) {
    const float* x = (const float*)d_in[0];       // [B,S,D]
    const float* w_qkv = (const float*)d_in[1];   // [3D,D]
    const float* b_qkv = (const float*)d_in[2];   // [3D]
    const float* w_out = (const float*)d_in[3];   // [D,D]
    const float* b_out = (const float*)d_in[4];   // [D]
    float* out = (float*)d_out;

    char* ws = (char*)d_ws;
    __bf16* xb = (__bf16*)(ws);                       // 4M elems,  8 MiB
    __bf16* wqkvb = (__bf16*)(ws + 8388608);          // 3M elems,  6 MiB
    __bf16* woutb = (__bf16*)(ws + 14680064);         // 1M elems,  2 MiB
    __bf16* Qb = (__bf16*)(ws + 16777216);            // 4M elems,  8 MiB
    __bf16* Kb = (__bf16*)(ws + 25165824);
    __bf16* Vb = (__bf16*)(ws + 33554432);
    __bf16* attnb = (__bf16*)(ws + 41943040);         // 4M elems,  8 MiB

    cast_f32_bf16<<<4096, 256, 0, stream>>>(x, xb, MROWS * EMB);
    cast_f32_bf16<<<4096, 256, 0, stream>>>(w_qkv, wqkvb, 3 * EMB * EMB);
    cast_f32_bf16<<<2048, 256, 0, stream>>>(w_out, woutb, EMB * EMB);

    // QKV projection: M=4096, N=3072 -> grid 64*24
    gemm_bf16<0><<<(MROWS / 64) * (3 * EMB / 128), 256, 0, stream>>>(
        xb, wqkvb, b_qkv, nullptr, Qb, Kb, Vb, 3 * EMB);

    // attention: B*H*(S/128) = 512 blocks
    attn_fwd<<<BATCH * NUM_HEADS * (SEQ / 128), 256, 0, stream>>>(Qb, Kb, Vb, attnb);

    // output projection: M=4096, N=1024 -> grid 64*8
    gemm_bf16<1><<<(MROWS / 64) * (EMB / 128), 256, 0, stream>>>(
        attnb, woutb, b_out, out, nullptr, nullptr, nullptr, EMB);
}
